// MemoryOptimizedMLA_63324997812258
// MI455X (gfx1250) — compile-verified, matrix operand order checked by reading, not run on hardware
//
#include <hip/hip_runtime.h>
#include <hip/hip_bf16.h>

// ---------------------------------------------------------------------------
// Types for CDNA5 WMMA (wave32): v_wmma_f32_16x16x32_bf16
// ---------------------------------------------------------------------------
typedef __bf16 bf16_t;
typedef __attribute__((ext_vector_type(16))) __bf16 bf16x16;
typedef __attribute__((ext_vector_type(8)))  __bf16 bf16x8;
typedef __attribute__((ext_vector_type(8)))  float  f32x8;

#define BH_TOKENS 4096   // B*S
#define SEQ 2048
#define DMODEL 5120
#define QA_RANK 1536
#define KVA_RANK 512
#define KPE 64
#define NHEADS 8
#define QHEAD 192
#define VHEAD 128

__device__ __forceinline__ f32x8 wmma_bf16(bf16x16 a, bf16x16 b, f32x8 c) {
    return __builtin_amdgcn_wmma_f32_16x16x32_bf16(
        false, a, false, b, (short)0, c, false, false);
}

// ---------------------------------------------------------------------------
// CDNA5 async global->LDS (ASYNCcnt-tracked DMA, no VGPR staging).
// dsaddr operand is a wave-relative LDS byte offset in a VGPR.
// ---------------------------------------------------------------------------
typedef __attribute__((address_space(3))) void lds_void;

__device__ __forceinline__ unsigned lds_off(const void* p) {
    return (unsigned)(unsigned long long)(lds_void*)p;  // addrspacecast -> 32-bit LDS offset
}
__device__ __forceinline__ void async_b128(unsigned dsoff, const void* gptr) {
    asm volatile("global_load_async_to_lds_b128 %0, %1, off"
                 :: "v"(dsoff), "v"(gptr) : "memory");
}
__device__ __forceinline__ void wait_async0() {
    asm volatile("s_wait_asynccnt 0" ::: "memory");
}

// Load a 16x32 bf16 fragment from an LDS tile stored [row][K] with padded
// stride (ISA 7.12.2: lane = M%16 / N%16, half = lane>=16, VGPRs 0..3 hold
// K = half*8..+7, VGPRs 4..7 hold K = 16+half*8..+7) -> two ds_load_b128.
__device__ __forceinline__ bf16x16 load_frag_row(const bf16_t* base, int stride,
                                                 int row0, int lane) {
    int m = lane & 15, half = lane >> 4;
    const bf16_t* p = base + (size_t)(row0 + m) * stride + half * 8;
    bf16x8 lo = *(const bf16x8*)(p);
    bf16x8 hi = *(const bf16x8*)(p + 16);
    bf16x16 r;
#pragma unroll
    for (int i = 0; i < 8; ++i) { r[i] = lo[i]; r[i + 8] = hi[i]; }
    return r;
}

// ---------------------------------------------------------------------------
// fp32 -> bf16 cast (grid-stride, float4 loads)
// ---------------------------------------------------------------------------
__global__ __launch_bounds__(256) void cast_f32_bf16(bf16_t* __restrict__ dst,
                                                     const float* __restrict__ src,
                                                     long n) {
    long i = ((long)blockIdx.x * 256 + threadIdx.x) * 4;
    long stride = (long)gridDim.x * 1024;
    for (; i + 3 < n; i += stride) {
        float4 v = *(const float4*)(src + i);
        dst[i + 0] = (bf16_t)v.x; dst[i + 1] = (bf16_t)v.y;
        dst[i + 2] = (bf16_t)v.z; dst[i + 3] = (bf16_t)v.w;
    }
}

// ---------------------------------------------------------------------------
// Weight cast + transpose: W[K][N] f32 -> WT[N][K] bf16.  K,N multiples of 64.
// LDS-staged so both global reads and writes are wide/coalesced.
// ---------------------------------------------------------------------------
__global__ __launch_bounds__(256) void cast_transpose_bf16(
    bf16_t* __restrict__ WT, const float* __restrict__ W, int K, int N) {
    __shared__ __align__(16) bf16_t st[64][72];  // [k][n], padded
    int kt = blockIdx.y * 64, nt = blockIdx.x * 64;
    {
        int r = threadIdx.x >> 2, c = (threadIdx.x & 3) * 16;
        const float* src = W + (size_t)(kt + r) * N + nt + c;
#pragma unroll
        for (int i = 0; i < 4; ++i) {
            float4 v = *(const float4*)(src + i * 4);
            st[r][c + i * 4 + 0] = (bf16_t)v.x; st[r][c + i * 4 + 1] = (bf16_t)v.y;
            st[r][c + i * 4 + 2] = (bf16_t)v.z; st[r][c + i * 4 + 3] = (bf16_t)v.w;
        }
    }
    __syncthreads();
    {
        int n = threadIdx.x >> 2, k4 = (threadIdx.x & 3) * 16;
        bf16x8 o0, o1;
#pragma unroll
        for (int i = 0; i < 8; ++i) { o0[i] = st[k4 + i][n]; o1[i] = st[k4 + 8 + i][n]; }
        bf16_t* dst = WT + (size_t)(nt + n) * K + kt + k4;
        *(bf16x8*)(dst)     = o0;
        *(bf16x8*)(dst + 8) = o1;
    }
}

// ---------------------------------------------------------------------------
// bf16 WMMA GEMM with pre-transposed B:  C[M,N] f32 = A[M,K] * BT[N,K]^T
// Block tile 128 x (WN*32), K-step 32.  Double-buffered LDS fed by async DMA.
// 8 waves as 4x2: wave = 32 rows x (WN*16) cols = 2*WN accumulators,
// 2*WN WMMAs per k-step.
// ---------------------------------------------------------------------------
template <int WN>  // 4 -> BN=128, 2 -> BN=64
__global__ __launch_bounds__(256) void gemm_bf16_wmma(
    float* __restrict__ C, const bf16_t* __restrict__ A,
    const bf16_t* __restrict__ BT, int M, int N, int K) {
    constexpr int BM = 128;
    constexpr int BN = WN * 32;
    constexpr int LD = 40;  // padded K-stride (80B: 16B-aligned, bank-spread)
    __shared__ __align__(16) bf16_t sA[2][BM * LD];
    __shared__ __align__(16) bf16_t sB[2][BN * LD];

    int tid = threadIdx.x, lane = tid & 31, w = tid >> 5;
    int wr = w >> 1, wc = w & 1;
    int rowBase = blockIdx.y * BM;
    int colBase = blockIdx.x * BN;

    int aR = tid >> 1, aC = (tid & 1) * 16;  // A: 128x32, 16 elems/thread

    auto issue_tile = [&](int buf, int kk) {
        const bf16_t* ga = A + (size_t)(rowBase + aR) * K + kk + aC;
        async_b128(lds_off(&sA[buf][aR * LD + aC]), ga);
        async_b128(lds_off(&sA[buf][aR * LD + aC + 8]), ga + 8);
        if constexpr (BN == 128) {
            int bR = tid >> 1, bC = (tid & 1) * 16;
            const bf16_t* gb = BT + (size_t)(colBase + bR) * K + kk + bC;
            async_b128(lds_off(&sB[buf][bR * LD + bC]), gb);
            async_b128(lds_off(&sB[buf][bR * LD + bC + 8]), gb + 8);
        } else {
            int bR = tid >> 2, bC = (tid & 3) * 8;
            const bf16_t* gb = BT + (size_t)(colBase + bR) * K + kk + bC;
            async_b128(lds_off(&sB[buf][bR * LD + bC]), gb);
        }
    };

    f32x8 acc[2][WN];
#pragma unroll
    for (int mi = 0; mi < 2; ++mi)
#pragma unroll
        for (int j = 0; j < WN; ++j) acc[mi][j] = (f32x8){};

    issue_tile(0, 0);
    int nk = K / 32;
    for (int t = 0; t < nk; ++t) {
        int cur = t & 1;
        wait_async0();
        __syncthreads();                       // tile t resident for all waves
        if (t + 1 < nk) issue_tile(cur ^ 1, (t + 1) * 32);  // DMA overlaps WMMA
        bf16x16 a0 = load_frag_row(&sA[cur][0], LD, wr * 32, lane);
        bf16x16 a1 = load_frag_row(&sA[cur][0], LD, wr * 32 + 16, lane);
#pragma unroll
        for (int j = 0; j < WN; ++j) {
            bf16x16 bf = load_frag_row(&sB[cur][0], LD, wc * (WN * 16) + j * 16, lane);
            acc[0][j] = wmma_bf16(a0, bf, acc[0][j]);
            acc[1][j] = wmma_bf16(a1, bf, acc[1][j]);
        }
        __syncthreads();                       // all readers done with tile t
    }

    int n = lane & 15, half = lane >> 4;
#pragma unroll
    for (int mi = 0; mi < 2; ++mi)
#pragma unroll
        for (int j = 0; j < WN; ++j)
#pragma unroll
            for (int r = 0; r < 8; ++r) {
                int mrow = rowBase + wr * 32 + mi * 16 + r + 8 * half;
                C[(size_t)mrow * N + colBase + wc * (WN * 16) + j * 16 + n] = acc[mi][j][r];
            }
}

// ---------------------------------------------------------------------------
// RMS-norm over `width` elems of each `in_stride`-wide row, * g, cast bf16.
// ---------------------------------------------------------------------------
__global__ __launch_bounds__(256) void rmsnorm_cast(
    bf16_t* __restrict__ out, const float* __restrict__ in,
    const float* __restrict__ g, int width, int in_stride) {
    __shared__ float red[8];
    int row = blockIdx.x;
    const float* x = in + (size_t)row * in_stride;
    float ss = 0.f;
    for (int i = threadIdx.x; i < width; i += 256) { float v = x[i]; ss = fmaf(v, v, ss); }
#pragma unroll
    for (int off = 16; off; off >>= 1) ss += __shfl_xor(ss, off, 32);
    if ((threadIdx.x & 31) == 0) red[threadIdx.x >> 5] = ss;
    __syncthreads();
    float tot = 0.f;
#pragma unroll
    for (int i = 0; i < 8; ++i) tot += red[i];
    float r = rsqrtf(tot / (float)width + 1e-6f);
    bf16_t* o = out + (size_t)row * width;
    for (int i = threadIdx.x; i < width; i += 256) o[i] = (bf16_t)(x[i] * r * g[i]);
}

// ---------------------------------------------------------------------------
// RoPE + pack kernels
// ---------------------------------------------------------------------------
__device__ __forceinline__ float rope_angle(float t, int f) {
    return t * __expf(-(float)(2 * f) * (9.2103404f / 64.0f));  // 10000^(-2f/64)
}

__global__ __launch_bounds__(256) void pack_q_rope(bf16_t* __restrict__ Qbf,
                                                   const float* __restrict__ qraw) {
    int token = blockIdx.x;
    int b = token >> 11, s = token & 2047;
    float t = (float)s / 40.0f;
    for (int idx = threadIdx.x; idx < NHEADS * QHEAD; idx += 256) {
        int h = idx / QHEAD, d = idx - h * QHEAD;
        const float* src = qraw + (size_t)token * (NHEADS * QHEAD) + h * QHEAD;
        float v;
        if (d < 128) {
            v = src[d];
        } else {
            int j = d - 128, f = j & 31;
            float ang = rope_angle(t, f);
            float c = __cosf(ang), sn = __sinf(ang);
            float x = src[d];
            v = (j < 32) ? (x * c - src[d + 32] * sn) : (x * c + src[d - 32] * sn);
        }
        Qbf[((size_t)(b * NHEADS + h) * SEQ + s) * QHEAD + d] = (bf16_t)v;
    }
}

__global__ __launch_bounds__(256) void pack_k_rope(bf16_t* __restrict__ Kbf,
                                                   const float* __restrict__ kvraw,
                                                   const float* __restrict__ kva_raw) {
    int token = blockIdx.x;
    int b = token >> 11, s = token & 2047;
    float t = (float)s / 40.0f;
    const float* kv = kvraw + (size_t)token * (NHEADS * 256);
    const float* pe = kva_raw + (size_t)token * (KVA_RANK + KPE) + KVA_RANK;
    for (int idx = threadIdx.x; idx < NHEADS * QHEAD; idx += 256) {
        int h = idx / QHEAD, d = idx - h * QHEAD;
        float v;
        if (d < 128) {
            v = kv[h * 256 + d];
        } else {
            int j = d - 128, f = j & 31;
            float ang = rope_angle(t, f);
            float c = __cosf(ang), sn = __sinf(ang);
            float x = pe[j];
            v = (j < 32) ? (x * c - pe[j + 32] * sn) : (x * c + pe[j - 32] * sn);
        }
        Kbf[((size_t)(b * NHEADS + h) * SEQ + s) * QHEAD + d] = (bf16_t)v;
    }
}

// V from kv_raw[token][h*256+128+d] -> Vt[bh][d][s] (transposed for PV WMMA),
// LDS-staged 64x64 tiles.  grid: (SEQ/64, 16 bh * 2 d-tiles)
__global__ __launch_bounds__(256) void pack_v_transpose(
    bf16_t* __restrict__ Vt, const float* __restrict__ kvraw) {
    __shared__ __align__(16) bf16_t st[64][72];  // [s][d]
    int bh = blockIdx.y >> 1, dt = (blockIdx.y & 1) * 64;
    int b = bh >> 3, h = bh & 7;
    int s0 = blockIdx.x * 64;
    {
        int r = threadIdx.x >> 2, c = (threadIdx.x & 3) * 16;
        const float* src = kvraw + (size_t)(b * SEQ + s0 + r) * (NHEADS * 256)
                         + h * 256 + 128 + dt + c;
#pragma unroll
        for (int i = 0; i < 4; ++i) {
            float4 v = *(const float4*)(src + i * 4);
            st[r][c + i * 4 + 0] = (bf16_t)v.x; st[r][c + i * 4 + 1] = (bf16_t)v.y;
            st[r][c + i * 4 + 2] = (bf16_t)v.z; st[r][c + i * 4 + 3] = (bf16_t)v.w;
        }
    }
    __syncthreads();
    {
        int d = threadIdx.x >> 2, k4 = (threadIdx.x & 3) * 16;
        bf16x8 o0, o1;
#pragma unroll
        for (int i = 0; i < 8; ++i) { o0[i] = st[k4 + i][d]; o1[i] = st[k4 + 8 + i][d]; }
        bf16_t* dst = Vt + ((size_t)bh * VHEAD + dt + d) * SEQ + s0 + k4;
        *(bf16x8*)(dst)     = o0;
        *(bf16x8*)(dst + 8) = o1;
    }
}

// ---------------------------------------------------------------------------
// Flash attention: 64 q-rows/block, 32-key tiles, online softmax.
// All tile loads via async DMA to LDS.  V pre-transposed: Vt[bh][d][s].
// ---------------------------------------------------------------------------
#define FA_QT 64
#define FA_KT 32
#define LDQ 200
#define LDK 200
#define LDVT 40
#define LDS_S 36
#define LDP 40

__global__ __launch_bounds__(256) void mla_flash_attn(
    bf16_t* __restrict__ O, const bf16_t* __restrict__ Q,
    const bf16_t* __restrict__ Km, const bf16_t* __restrict__ Vt) {
    __shared__ __align__(16) bf16_t sQ[FA_QT * LDQ];
    __shared__ __align__(16) bf16_t sK[FA_KT * LDK];
    __shared__ __align__(16) bf16_t sVT[VHEAD * LDVT];  // [v-dim][key]
    __shared__ __align__(16) float  sS[FA_QT * LDS_S];
    __shared__ __align__(16) bf16_t sP[FA_QT * LDP];
    __shared__ float m_run[FA_QT], l_run[FA_QT], rscale[FA_QT];

    int tid = threadIdx.x, lane = tid & 31, w = tid >> 5;
    int wr = w >> 1, wc = w & 1;
    int bh = blockIdx.y;
    int qBase = blockIdx.x * FA_QT;
    const bf16_t* Qp  = Q  + (size_t)bh * SEQ * QHEAD;
    const bf16_t* Kp  = Km + (size_t)bh * SEQ * QHEAD;
    const bf16_t* Vtp = Vt + (size_t)bh * VHEAD * SEQ;

    { // Q tile 64x192: 6 async b128 per thread
        int r = tid >> 2, c0 = (tid & 3) * 48;
#pragma unroll
        for (int i = 0; i < 6; ++i)
            async_b128(lds_off(&sQ[r * LDQ + c0 + i * 8]),
                       Qp + (size_t)(qBase + r) * QHEAD + c0 + i * 8);
    }
    if (tid < FA_QT) { m_run[tid] = -1e30f; l_run[tid] = 0.f; }

    f32x8 o[4] = {{}, {}, {}, {}};
    const float scale = 0.07216878365f;  // 1/sqrt(192)

    for (int kt = 0; kt < SEQ; kt += FA_KT) {
        { // K tile 32x192
            int r = tid >> 3, c0 = (tid & 7) * 24;
#pragma unroll
            for (int i = 0; i < 3; ++i)
                async_b128(lds_off(&sK[r * LDK + c0 + i * 8]),
                           Kp + (size_t)(kt + r) * QHEAD + c0 + i * 8);
        }
        { // VT tile 128x32 (already [d][s] in memory)
            int d = tid >> 1, c0 = (tid & 1) * 16;
            async_b128(lds_off(&sVT[d * LDVT + c0]), Vtp + (size_t)d * SEQ + kt + c0);
            async_b128(lds_off(&sVT[d * LDVT + c0 + 8]), Vtp + (size_t)d * SEQ + kt + c0 + 8);
        }
        wait_async0();
        __syncthreads();

        // S = Q K^T (64x32); K-dim 192 = 6 WMMA steps
        f32x8 sacc = {};
#pragma unroll
        for (int dk = 0; dk < QHEAD; dk += 32) {
            bf16x16 aq = load_frag_row(sQ + dk, LDQ, wr * 16, lane);
            bf16x16 bk = load_frag_row(sK + dk, LDK, wc * 16, lane);
            sacc = wmma_bf16(aq, bk, sacc);
        }
        {
            int n = lane & 15, half = lane >> 4;
#pragma unroll
            for (int r = 0; r < 8; ++r)
                sS[(wr * 16 + r + 8 * half) * LDS_S + wc * 16 + n] = sacc[r] * scale;
        }
        __syncthreads();

        // online softmax: thread t < 64 owns q-row t
        if (tid < FA_QT) {
            float mold = m_run[tid], mnew = mold;
#pragma unroll
            for (int c = 0; c < FA_KT; ++c) mnew = fmaxf(mnew, sS[tid * LDS_S + c]);
            float corr = __expf(mold - mnew);
            float lsum = 0.f;
#pragma unroll
            for (int c = 0; c < FA_KT; ++c) {
                float p = __expf(sS[tid * LDS_S + c] - mnew);
                lsum += p;
                sP[tid * LDP + c] = (bf16_t)p;
            }
            m_run[tid] = mnew;
            l_run[tid] = l_run[tid] * corr + lsum;
            rscale[tid] = corr;
        }
        __syncthreads();

        // rescale O, then O += P * V (one 32-deep WMMA step)
        {
            int half = lane >> 4;
#pragma unroll
            for (int r = 0; r < 8; ++r) {
                float corr = rscale[wr * 16 + r + 8 * half];
#pragma unroll
                for (int a = 0; a < 4; ++a) o[a][r] *= corr;
            }
            bf16x16 ap = load_frag_row(sP, LDP, wr * 16, lane);
#pragma unroll
            for (int a = 0; a < 4; ++a) {
                bf16x16 bv = load_frag_row(sVT, LDVT, wc * 64 + a * 16, lane);
                o[a] = wmma_bf16(ap, bv, o[a]);
            }
        }
        __syncthreads();
    }

    int b = bh >> 3, h = bh & 7;
    int n = lane & 15, half = lane >> 4;
#pragma unroll
    for (int r = 0; r < 8; ++r) {
        int qrow = qBase + wr * 16 + r + 8 * half;
        float inv = 1.f / l_run[wr * 16 + r + 8 * half];
        size_t orow = ((size_t)b * SEQ + qrow) * (NHEADS * VHEAD) + h * VHEAD;
#pragma unroll
        for (int a = 0; a < 4; ++a)
            O[orow + wc * 64 + a * 16 + n] = (bf16_t)(o[a][r] * inv);
    }
}

// ---------------------------------------------------------------------------
// Host launcher
// ---------------------------------------------------------------------------
extern "C" void kernel_launch(void* const* d_in, const int* in_sizes, int n_in,
                              void* d_out, int out_size, void* d_ws, size_t ws_size,
                              hipStream_t stream) {
    (void)in_sizes; (void)n_in; (void)out_size; (void)ws_size;
    const float* h    = (const float*)d_in[0];
    const float* Wqa  = (const float*)d_in[1];
    const float* gqa  = (const float*)d_in[2];
    const float* Wqb  = (const float*)d_in[3];
    const float* Wkva = (const float*)d_in[4];
    const float* gkv  = (const float*)d_in[5];
    const float* Wkvb = (const float*)d_in[6];
    const float* Wout = (const float*)d_in[7];
    float* out = (float*)d_out;

    char* ws = (char*)d_ws;
    size_t off = 0;
    auto alloc = [&](size_t bytes) { size_t o = off; off += (bytes + 255) & ~(size_t)255; return o; };

    bf16_t* h_bf    = (bf16_t*)(ws + alloc((size_t)BH_TOKENS * DMODEL * 2));  // reused: kv_raw f32
    bf16_t* WTqa    = (bf16_t*)(ws + alloc((size_t)DMODEL * QA_RANK * 2));    // [1536][5120]
    bf16_t* WTqb    = (bf16_t*)(ws + alloc((size_t)QA_RANK * NHEADS * QHEAD * 2)); // [1536][1536]
    bf16_t* WTkva   = (bf16_t*)(ws + alloc((size_t)DMODEL * (KVA_RANK + KPE) * 2)); // [576][5120]
    bf16_t* WTkvb   = (bf16_t*)(ws + alloc((size_t)KVA_RANK * NHEADS * 256 * 2));   // [2048][512]
    bf16_t* WTout   = (bf16_t*)(ws + alloc((size_t)NHEADS * VHEAD * DMODEL * 2));   // [5120][1024]
    float*  qa_raw  = (float*)(ws + alloc((size_t)BH_TOKENS * QA_RANK * 4));  // reused: q_raw
    float*  kva_raw = (float*)(ws + alloc((size_t)BH_TOKENS * (KVA_RANK + KPE) * 4));
    bf16_t* qa_bf   = (bf16_t*)(ws + alloc((size_t)BH_TOKENS * QA_RANK * 2));
    bf16_t* kvn_bf  = (bf16_t*)(ws + alloc((size_t)BH_TOKENS * KVA_RANK * 2));
    bf16_t* Qbf     = (bf16_t*)(ws + alloc((size_t)BH_TOKENS * NHEADS * QHEAD * 2));
    bf16_t* Kbf     = (bf16_t*)(ws + alloc((size_t)BH_TOKENS * NHEADS * QHEAD * 2));
    bf16_t* Vt      = (bf16_t*)(ws + alloc((size_t)BH_TOKENS * NHEADS * VHEAD * 2));
    bf16_t* attn_bf = (bf16_t*)(ws + alloc((size_t)BH_TOKENS * NHEADS * VHEAD * 2));
    float*  q_raw   = qa_raw;        // alias: qa_raw dead after rmsnorm
    float*  kv_raw  = (float*)h_bf;  // alias: h_bf dead after first two GEMMs

    // 1) casts: activations plain, weights cast+transpose to [N][K]
    cast_f32_bf16<<<2048, 256, 0, stream>>>(h_bf, h, (long)BH_TOKENS * DMODEL);
    cast_transpose_bf16<<<dim3(QA_RANK / 64, DMODEL / 64), 256, 0, stream>>>(
        WTqa, Wqa, DMODEL, QA_RANK);
    cast_transpose_bf16<<<dim3((NHEADS * QHEAD) / 64, QA_RANK / 64), 256, 0, stream>>>(
        WTqb, Wqb, QA_RANK, NHEADS * QHEAD);
    cast_transpose_bf16<<<dim3((KVA_RANK + KPE) / 64, DMODEL / 64), 256, 0, stream>>>(
        WTkva, Wkva, DMODEL, KVA_RANK + KPE);
    cast_transpose_bf16<<<dim3((NHEADS * 256) / 64, KVA_RANK / 64), 256, 0, stream>>>(
        WTkvb, Wkvb, KVA_RANK, NHEADS * 256);
    cast_transpose_bf16<<<dim3(DMODEL / 64, (NHEADS * VHEAD) / 64), 256, 0, stream>>>(
        WTout, Wout, NHEADS * VHEAD, DMODEL);

    // 2) q_a = h @ W_qa ; kv_a = h @ W_kva
    gemm_bf16_wmma<4><<<dim3(QA_RANK / 128, BH_TOKENS / 128), 256, 0, stream>>>(
        qa_raw, h_bf, WTqa, BH_TOKENS, QA_RANK, DMODEL);
    gemm_bf16_wmma<2><<<dim3((KVA_RANK + KPE) / 64, BH_TOKENS / 128), 256, 0, stream>>>(
        kva_raw, h_bf, WTkva, BH_TOKENS, KVA_RANK + KPE, DMODEL);

    // 3) rms norms
    rmsnorm_cast<<<BH_TOKENS, 256, 0, stream>>>(qa_bf, qa_raw, gqa, QA_RANK, QA_RANK);
    rmsnorm_cast<<<BH_TOKENS, 256, 0, stream>>>(kvn_bf, kva_raw, gkv, KVA_RANK, KVA_RANK + KPE);

    // 4) q = q_a_norm @ W_qb ; kv = kv_norm @ W_kvb
    gemm_bf16_wmma<4><<<dim3((NHEADS * QHEAD) / 128, BH_TOKENS / 128), 256, 0, stream>>>(
        q_raw, qa_bf, WTqb, BH_TOKENS, NHEADS * QHEAD, QA_RANK);
    gemm_bf16_wmma<4><<<dim3((NHEADS * 256) / 128, BH_TOKENS / 128), 256, 0, stream>>>(
        kv_raw, kvn_bf, WTkvb, BH_TOKENS, NHEADS * 256, KVA_RANK);

    // 5) RoPE + pack
    pack_q_rope<<<BH_TOKENS, 256, 0, stream>>>(Qbf, q_raw);
    pack_k_rope<<<BH_TOKENS, 256, 0, stream>>>(Kbf, kv_raw, kva_raw);
    pack_v_transpose<<<dim3(SEQ / 64, 2 * NHEADS * 2), 256, 0, stream>>>(Vt, kv_raw);

    // 6) flash attention
    mla_flash_attn<<<dim3(SEQ / FA_QT, 2 * NHEADS), 256, 0, stream>>>(attn_bf, Qbf, Kbf, Vt);

    // 7) out = attn @ W_out
    gemm_bf16_wmma<4><<<dim3(DMODEL / 128, BH_TOKENS / 128), 256, 0, stream>>>(
        out, attn_bf, WTout, BH_TOKENS, DMODEL, NHEADS * VHEAD);
}
